// MultiHeadLatentAttention_74019466379279
// MI455X (gfx1250) — matrix operand order repaired and checked by reading, not compiled, not run on hardware
//
#include <hip/hip_runtime.h>
#include <math.h>

// ---------------------------------------------------------------------------
// MLA forward for MI455X (gfx1250, wave32, WMMA).
// fp32->bf16 convert, 7 WMMA GEMM projections (128x64 block tile, 32x32 wave
// tile, double-buffered async global->LDS copies), RoPE, flash attention
// (WMMA QK^T and PV, online softmax with DPP reductions, double-buffered
// K/V tiles), final WMMA GEMM -> fp32.
// ---------------------------------------------------------------------------

typedef __bf16 bf16;
typedef __attribute__((ext_vector_type(4)))  float f32x4;
typedef __attribute__((ext_vector_type(8)))  float f32x8;
typedef __attribute__((ext_vector_type(8)))  bf16  bf16x8;
typedef __attribute__((ext_vector_type(16))) bf16  bf16x16;

#define HIDDEN   2048
#define HEADS    16
#define HEAD_DIM 128
#define HALFD    64
#define LATENT   512
#define SEQ      2048
#define SCALE    0.08838834764831845f   // 1/sqrt(128)

// -------------------- async global->LDS (CDNA5 ASYNC path) ------------------

__device__ __forceinline__ void async_b128(const bf16* g, const bf16* l) {
  // GLOBAL_LOAD_ASYNC_TO_LDS_B128, GV mode: vdst = LDS byte offset (low 32
  // bits of the generic shared pointer), vaddr = 64-bit global address.
  asm volatile("global_load_async_to_lds_b128 %0, %1, off"
               :: "v"((unsigned int)(unsigned long long)l), "v"(g)
               : "memory");
}

__device__ __forceinline__ void wait_async() {
  asm volatile("s_wait_asynccnt 0x0" ::: "memory");
}

// -------------------- fragment helpers (wave32 WMMA layouts) ----------------

__device__ __forceinline__ bf16x16 frag_cat(bf16x8 lo, bf16x8 hi) {
  return __builtin_shufflevector(lo, hi, 0,1,2,3,4,5,6,7,8,9,10,11,12,13,14,15);
}

// A-matrix 16x32 bf16 fragment. `base` points at this lane's row, K=0.
__device__ __forceinline__ bf16x16 load_a_frag(const bf16* base, int lane) {
  int o = (lane & 16) ? 8 : 0;
  bf16x8 lo = *(const bf16x8*)(base + o);
  bf16x8 hi = *(const bf16x8*)(base + o + 16);
  return frag_cat(lo, hi);
}

// B-matrix 32x16 bf16 fragment from B^T storage (contiguous K per column).
__device__ __forceinline__ bf16x16 load_b_frag(const bf16* col, int lane) {
  int o = (lane & 16) ? 8 : 0;
  bf16x8 lo = *(const bf16x8*)(col + o);
  bf16x8 hi = *(const bf16x8*)(col + o + 16);
  return frag_cat(lo, hi);
}

__device__ __forceinline__ f32x8 wmma_bf16(bf16x16 a, bf16x16 b, f32x8 c) {
  return __builtin_amdgcn_wmma_f32_16x16x32_bf16(false, a, false, b,
                                                 (short)0, c, false, false);
}

// -------------------- 16-lane reductions (DPP butterfly, VALU-only) ---------

#if __has_builtin(__builtin_amdgcn_update_dpp)
#define MLA_HAVE_DPP 1
template <int CTRL>
__device__ __forceinline__ float dpp_f(float x) {
  int xi = __float_as_int(x);
  int r  = __builtin_amdgcn_update_dpp(xi, xi, CTRL, 0xf, 0xf, true);
  return __int_as_float(r);
}
#endif

__device__ __forceinline__ float red16_max(float x) {
#ifdef MLA_HAVE_DPP
  x = fmaxf(x, dpp_f<0xB1>(x));    // quad_perm [1,0,3,2]
  x = fmaxf(x, dpp_f<0x4E>(x));    // quad_perm [2,3,0,1]
  x = fmaxf(x, dpp_f<0x141>(x));   // row_half_mirror
  x = fmaxf(x, dpp_f<0x140>(x));   // row_mirror
#else
  x = fmaxf(x, __shfl_xor(x, 1, 32));
  x = fmaxf(x, __shfl_xor(x, 2, 32));
  x = fmaxf(x, __shfl_xor(x, 4, 32));
  x = fmaxf(x, __shfl_xor(x, 8, 32));
#endif
  return x;
}

__device__ __forceinline__ float red16_sum(float x) {
#ifdef MLA_HAVE_DPP
  x += dpp_f<0xB1>(x);
  x += dpp_f<0x4E>(x);
  x += dpp_f<0x141>(x);
  x += dpp_f<0x140>(x);
#else
  x += __shfl_xor(x, 1, 32);
  x += __shfl_xor(x, 2, 32);
  x += __shfl_xor(x, 4, 32);
  x += __shfl_xor(x, 8, 32);
#endif
  return x;
}

__device__ __forceinline__ int uniform_i(int v) {
  return __builtin_amdgcn_readfirstlane(v);
}

// -------------------- fp32 -> bf16 conversion -------------------------------

__global__ void __launch_bounds__(256)
cvt_f32_bf16(const float* __restrict__ src, bf16* __restrict__ dst, int n4) {
  int i = blockIdx.x * 256 + threadIdx.x;
  if (i < n4) {
    f32x4 f = ((const f32x4*)src)[i];
    bf16* p = dst + (size_t)i * 4;
    p[0] = (bf16)f.x; p[1] = (bf16)f.y; p[2] = (bf16)f.z; p[3] = (bf16)f.w;
  }
}

// -------------------- bf16 WMMA GEMM (double buffered) ----------------------
// C[M,N] = A[M,K] * B[K,N], bf16 row-major, fp32 accumulate.
// Block: 256 threads = 8 waves; tile 128x64, K-step 32.
// Wave w: rows (w/2)*32, cols (w%2)*32 -> 2x2 of 16x16 WMMA tiles.

#define GSTR 40   // LDS row stride in halfs (80 bytes, 16B aligned)

template <typename OutT>
__global__ void __launch_bounds__(256)
gemm_bf16_wmma(const bf16* __restrict__ A, const bf16* __restrict__ B,
               OutT* __restrict__ C, int M, int N, int K) {
  __shared__ bf16 As[2][128 * GSTR];
  __shared__ bf16 Bt[2][64 * GSTR];

  const int tid  = threadIdx.x;
  const int lane = tid & 31;
  const int w    = tid >> 5;
  const int m0   = blockIdx.x * 128;
  const int n0   = blockIdx.y * 64;
  const int wm   = (w >> 1) * 32;
  const int wn   = (w & 1) * 32;

  const int alr = tid >> 1, alc = (tid & 1) * 16;   // A loader: row, 16-half col
  const int bkr = tid >> 3, bng = (tid & 7) * 8;    // B loader: k-row, n-group

  f32x8 c00 = {}, c01 = {}, c10 = {}, c11 = {};

  const int nk = K >> 5;

  // prologue: stage tile 0
  {
    const bf16* ga = A + (size_t)(m0 + alr) * K + alc;
    async_b128(ga,     &As[0][alr * GSTR + alc]);
    async_b128(ga + 8, &As[0][alr * GSTR + alc + 8]);
    bf16x8 bv = *(const bf16x8*)(B + (size_t)bkr * N + n0 + bng);
#pragma unroll
    for (int i = 0; i < 8; ++i)
      Bt[0][(bng + i) * GSTR + bkr] = bv[i];
  }
  wait_async();
  __syncthreads();

  for (int kt = 0; kt < nk; ++kt) {
    const int cur  = kt & 1;
    const int nxt  = cur ^ 1;
    const int more = uniform_i(kt + 1 < nk);

    bf16x8 bvn = {};
    if (more) {   // stage tile kt+1 while computing tile kt
      const int k1 = (kt + 1) << 5;
      const bf16* ga = A + (size_t)(m0 + alr) * K + k1 + alc;
      async_b128(ga,     &As[nxt][alr * GSTR + alc]);
      async_b128(ga + 8, &As[nxt][alr * GSTR + alc + 8]);
      bvn = *(const bf16x8*)(B + (size_t)(k1 + bkr) * N + n0 + bng);
    }

    const int row0 = wm + (lane & 15);
    bf16x16 a0 = load_a_frag(&As[cur][row0 * GSTR], lane);
    bf16x16 a1 = load_a_frag(&As[cur][(row0 + 16) * GSTR], lane);
    bf16x16 b0 = load_b_frag(&Bt[cur][(wn + (lane & 15)) * GSTR], lane);
    bf16x16 b1 = load_b_frag(&Bt[cur][(wn + 16 + (lane & 15)) * GSTR], lane);
    c00 = wmma_bf16(a0, b0, c00);
    c01 = wmma_bf16(a0, b1, c01);
    c10 = wmma_bf16(a1, b0, c10);
    c11 = wmma_bf16(a1, b1, c11);

    if (more) {
#pragma unroll
      for (int i = 0; i < 8; ++i)
        Bt[nxt][(bng + i) * GSTR + bkr] = bvn[i];
    }
    wait_async();
    __syncthreads();
  }

  // C layout: elem r -> M = r + 8*(lane>>4), N = lane&15
  const int mb = m0 + wm + ((lane >> 4) << 3);
  const int nb = n0 + wn + (lane & 15);
#pragma unroll
  for (int r = 0; r < 8; ++r) {
    C[(size_t)(mb + r) * N + nb]           = (OutT)c00[r];
    C[(size_t)(mb + r) * N + nb + 16]      = (OutT)c01[r];
    C[(size_t)(mb + 16 + r) * N + nb]      = (OutT)c10[r];
    C[(size_t)(mb + 16 + r) * N + nb + 16] = (OutT)c11[r];
  }
}

// -------------------- RoPE (in-place on bf16 [rows, H*64]) ------------------

__global__ void __launch_bounds__(256)
rope_kernel(bf16* __restrict__ t, int total) {
  int idx = blockIdx.x * 256 + threadIdx.x;   // one thread per (row, h, j<32)
  if (idx >= total) return;
  int j = idx & 31;
  int h = (idx >> 5) & (HEADS - 1);
  int row = idx >> 9;
  int s = row & (SEQ - 1);
  float inv = __powf(10000.0f, -(float)(2 * j) / (float)HALFD);
  float ang = (float)s * inv;
  float sn, cs;
  __sincosf(ang, &sn, &cs);
  bf16* p = t + (size_t)row * (HEADS * HALFD) + h * HALFD + j;
  float x1 = (float)p[0];
  float x2 = (float)p[32];
  p[0]  = (bf16)(x1 * cs - x2 * sn);
  p[32] = (bf16)(x2 * cs + x1 * sn);
}

// -------------------- flash attention (double buffered) ---------------------
// grid (S/128, HEADS, B), block 256 (8 waves). Wave w owns q rows
// [qt*128 + w*16, +16). Streams 32-key tiles with online softmax.

#define KSTR 72   // K tile LDS stride (64 + 8 pad, 144B, 16B aligned)
#define VSTR 40   // V^T / P tile LDS stride (32 + 8 pad, 80B, 16B aligned)

__global__ void __launch_bounds__(256)
mla_flash_attn(const bf16* __restrict__ qn, const bf16* __restrict__ qr,
               const bf16* __restrict__ kn, const bf16* __restrict__ kr,
               const bf16* __restrict__ v,  bf16* __restrict__ y) {
  __shared__ bf16 Kn[2][32 * KSTR];
  __shared__ bf16 Kr[2][32 * KSTR];
  __shared__ bf16 Vt[2][128 * VSTR];
  __shared__ bf16 Pl[8 * 16 * VSTR];

  const int qt = blockIdx.x, h = blockIdx.y, b = blockIdx.z;
  const int tid = threadIdx.x, lane = tid & 31, w = tid >> 5;

  const size_t grow_q = (size_t)b * SEQ + qt * 128 + w * 16 + (lane & 15);

  // Q fragments (2 nope + 2 rope chunks of 32 dims) live in registers
  bf16x16 aq[4];
  {
    const bf16* pn = qn + grow_q * (HEADS * HALFD) + h * HALFD;
    const bf16* pr = qr + grow_q * (HEADS * HALFD) + h * HALFD;
    aq[0] = load_a_frag(pn, lane);
    aq[1] = load_a_frag(pn + 32, lane);
    aq[2] = load_a_frag(pr, lane);
    aq[3] = load_a_frag(pr + 32, lane);
  }

  float m_i[8], l_i[8];
  f32x8 o[8];
#pragma unroll
  for (int r = 0; r < 8; ++r) { m_i[r] = -3.0e38f; l_i[r] = 0.0f; }
#pragma unroll
  for (int d = 0; d < 8; ++d) o[d] = (f32x8){};

  const int qmin_w  = qt * 128 + w * 16;       // first q row of wave
  const int qmax_w  = qmin_w + 15;             // last q row of wave
  const int qbase_c = qmin_w + ((lane >> 4) << 3);
  const int nsteps  = (qt + 1) * 4;

  const int lkr = tid >> 3, ldg = tid & 7;

  // prologue: stage key tile 0
  {
    const size_t krow = (size_t)b * SEQ + lkr;
    async_b128(kn + krow * (HEADS * HALFD) + h * HALFD + ldg * 8,
               &Kn[0][lkr * KSTR + ldg * 8]);
    async_b128(kr + krow * (HEADS * HALFD) + h * HALFD + ldg * 8,
               &Kr[0][lkr * KSTR + ldg * 8]);
    const bf16* vp = v + krow * HIDDEN + h * HEAD_DIM + ldg * 16;
    bf16x8 v0 = *(const bf16x8*)(vp);
    bf16x8 v1 = *(const bf16x8*)(vp + 8);
#pragma unroll
    for (int i = 0; i < 8; ++i) Vt[0][(ldg * 16 + i) * VSTR + lkr]     = v0[i];
#pragma unroll
    for (int i = 0; i < 8; ++i) Vt[0][(ldg * 16 + 8 + i) * VSTR + lkr] = v1[i];
  }
  wait_async();
  __syncthreads();

  for (int kt = 0; kt < nsteps; ++kt) {
    const int kb   = kt * 32;
    const int cur  = kt & 1;
    const int nxt  = cur ^ 1;
    const int more = uniform_i(kt + 1 < nsteps);

    bf16x8 v0n = {}, v1n = {};
    if (more) {   // stage key tile kt+1 while computing tile kt
      const size_t krow = (size_t)b * SEQ + kb + 32 + lkr;
      async_b128(kn + krow * (HEADS * HALFD) + h * HALFD + ldg * 8,
                 &Kn[nxt][lkr * KSTR + ldg * 8]);
      async_b128(kr + krow * (HEADS * HALFD) + h * HALFD + ldg * 8,
                 &Kr[nxt][lkr * KSTR + ldg * 8]);
      const bf16* vp = v + krow * HIDDEN + h * HEAD_DIM + ldg * 16;
      v0n = *(const bf16x8*)(vp);
      v1n = *(const bf16x8*)(vp + 8);
    }

    if (uniform_i(kb <= qmax_w)) {    // scalar-uniform: EXEC stays all ones
      f32x8 s0 = {}, s1 = {};
      const int c0 = (lane & 15) * KSTR;
      const int c1 = (16 + (lane & 15)) * KSTR;
#pragma unroll
      for (int c = 0; c < 2; ++c) {
        s0 = wmma_bf16(aq[c], load_b_frag(&Kn[cur][c0 + c * 32], lane), s0);
        s1 = wmma_bf16(aq[c], load_b_frag(&Kn[cur][c1 + c * 32], lane), s1);
      }
#pragma unroll
      for (int c = 0; c < 2; ++c) {
        s0 = wmma_bf16(aq[2 + c], load_b_frag(&Kr[cur][c0 + c * 32], lane), s0);
        s1 = wmma_bf16(aq[2 + c], load_b_frag(&Kr[cur][c1 + c * 32], lane), s1);
      }

      const int nm   = uniform_i(kb + 31 > qmin_w);   // diagonal step?
      const int key0 = kb + (lane & 15);
      const int key1 = key0 + 16;
#pragma unroll
      for (int r = 0; r < 8; ++r) {
        float x0 = s0[r] * SCALE;
        float x1 = s1[r] * SCALE;
        if (nm) {   // only the 1-2 diagonal steps pay the mask VALU cost
          const int qrow = qbase_c + r;
          if (key0 > qrow) x0 = -1.0e30f;
          if (key1 > qrow) x1 = -1.0e30f;
        }

        float rm = red16_max(fmaxf(x0, x1));

        float mnew  = fmaxf(m_i[r], rm);
        float alpha = __expf(m_i[r] - mnew);
        float p0 = __expf(x0 - mnew);
        float p1 = __expf(x1 - mnew);
        float rs = red16_sum(p0 + p1);
        l_i[r] = l_i[r] * alpha + rs;
        m_i[r] = mnew;
#pragma unroll
        for (int d = 0; d < 8; ++d) o[d][r] = o[d][r] * alpha;

        // P (C layout) -> wave-private LDS, row = r + 8*(lane>>4)
        const int prow = r + ((lane >> 4) << 3);
        Pl[w * 16 * VSTR + prow * VSTR + (lane & 15)]      = (bf16)p0;
        Pl[w * 16 * VSTR + prow * VSTR + 16 + (lane & 15)] = (bf16)p1;
      }

      asm volatile("s_wait_dscnt 0" ::: "memory");  // intra-wave LDS RAW

      bf16x16 ap = load_a_frag(&Pl[w * 16 * VSTR + (lane & 15) * VSTR], lane);
#pragma unroll
      for (int d = 0; d < 8; ++d) {
        bf16x16 bv = load_b_frag(&Vt[cur][(d * 16 + (lane & 15)) * VSTR], lane);
        o[d] = wmma_bf16(ap, bv, o[d]);
      }
    }

    if (more) {   // publish V^T for the next step (buffer no one reads now)
#pragma unroll
      for (int i = 0; i < 8; ++i) Vt[nxt][(ldg * 16 + i) * VSTR + lkr]     = v0n[i];
#pragma unroll
      for (int i = 0; i < 8; ++i) Vt[nxt][(ldg * 16 + 8 + i) * VSTR + lkr] = v1n[i];
    }
    wait_async();
    __syncthreads();
  }

  const size_t row0 = (size_t)b * SEQ + qt * 128 + w * 16 + ((lane >> 4) << 3);
#pragma unroll
  for (int r = 0; r < 8; ++r) {
    float invl = 1.0f / l_i[r];
    bf16* yp = y + (row0 + r) * HIDDEN + h * HEAD_DIM + (lane & 15);
#pragma unroll
    for (int d = 0; d < 8; ++d)
      yp[d * 16] = (bf16)(o[d][r] * invl);
  }
}

// -------------------- host orchestration ------------------------------------

extern "C" void kernel_launch(void* const* d_in, const int* in_sizes, int n_in,
                              void* d_out, int out_size, void* d_ws, size_t ws_size,
                              hipStream_t stream) {
  const float* x    = (const float*)d_in[0];
  const float* wkvd = (const float*)d_in[1];
  const float* wqd  = (const float*)d_in[2];
  const float* wku  = (const float*)d_in[3];
  const float* wqu  = (const float*)d_in[4];
  const float* wvu  = (const float*)d_in[5];
  const float* wrk  = (const float*)d_in[6];
  const float* wrq  = (const float*)d_in[7];
  const float* wo   = (const float*)d_in[8];
  float* out = (float*)d_out;

  const int rows = in_sizes[0] / HIDDEN;          // B * S
  const int B    = rows / SEQ;

  char* ws = (char*)d_ws;
  size_t off = 0;
  auto alloc = [&](size_t elems) { bf16* p = (bf16*)(ws + off); off += elems * 2; return p; };

  bf16* xb    = alloc((size_t)rows * HIDDEN);
  bf16* wkvdb = alloc((size_t)HIDDEN * LATENT);
  bf16* wqdb  = alloc((size_t)HIDDEN * LATENT);
  bf16* wkub  = alloc((size_t)LATENT * HEADS * HALFD);
  bf16* wqub  = alloc((size_t)LATENT * HEADS * HALFD);
  bf16* wvub  = alloc((size_t)LATENT * HIDDEN);
  bf16* wrkb  = alloc((size_t)HIDDEN * (HIDDEN / 2));
  bf16* wrqb  = alloc((size_t)LATENT * (HIDDEN / 2));
  bf16* wob   = alloc((size_t)HIDDEN * HIDDEN);
  bf16* kvlat = alloc((size_t)rows * LATENT);
  bf16* qlat  = alloc((size_t)rows * LATENT);
  bf16* knope = alloc((size_t)rows * HEADS * HALFD);
  bf16* qnope = alloc((size_t)rows * HEADS * HALFD);
  bf16* vbuf  = alloc((size_t)rows * HIDDEN);
  bf16* qrope = alloc((size_t)rows * HEADS * HALFD);
  bf16* krope = alloc((size_t)rows * HEADS * HALFD);
  bf16* ybuf  = alloc((size_t)rows * HIDDEN);

  auto cvt = [&](const float* s, bf16* d, int n) {
    int n4 = n / 4;
    cvt_f32_bf16<<<(n4 + 255) / 256, 256, 0, stream>>>(s, d, n4);
  };
  cvt(x,    xb,    in_sizes[0]);
  cvt(wkvd, wkvdb, in_sizes[1]);
  cvt(wqd,  wqdb,  in_sizes[2]);
  cvt(wku,  wkub,  in_sizes[3]);
  cvt(wqu,  wqub,  in_sizes[4]);
  cvt(wvu,  wvub,  in_sizes[5]);
  cvt(wrk,  wrkb,  in_sizes[6]);
  cvt(wrq,  wrqb,  in_sizes[7]);
  cvt(wo,   wob,   in_sizes[8]);

  auto gemm_b = [&](const bf16* A, const bf16* Bm, bf16* C, int M, int N, int K) {
    dim3 g(M / 128, N / 64);
    gemm_bf16_wmma<bf16><<<g, 256, 0, stream>>>(A, Bm, C, M, N, K);
  };

  gemm_b(xb,    wkvdb, kvlat, rows, LATENT, HIDDEN);             // kv_latent
  gemm_b(xb,    wqdb,  qlat,  rows, LATENT, HIDDEN);             // q_latent
  gemm_b(kvlat, wkub,  knope, rows, HEADS * HALFD, LATENT);      // k (nope)
  gemm_b(qlat,  wqub,  qnope, rows, HEADS * HALFD, LATENT);      // q (nope)
  gemm_b(kvlat, wvub,  vbuf,  rows, HIDDEN, LATENT);             // v
  gemm_b(qlat,  wrqb,  qrope, rows, HIDDEN / 2, LATENT);         // q_rope
  gemm_b(xb,    wrkb,  krope, rows, HIDDEN / 2, HIDDEN);         // k_rope

  const int rope_total = rows * HEADS * 32;
  rope_kernel<<<(rope_total + 255) / 256, 256, 0, stream>>>(qrope, rope_total);
  rope_kernel<<<(rope_total + 255) / 256, 256, 0, stream>>>(krope, rope_total);

  dim3 ag(SEQ / 128, HEADS, B);
  mla_flash_attn<<<ag, 256, 0, stream>>>(qnope, qrope, knope, krope, vbuf, ybuf);

  dim3 og(rows / 128, HIDDEN / 64);
  gemm_bf16_wmma<float><<<og, 256, 0, stream>>>(ybuf, wob, out, rows, HIDDEN, HIDDEN);
}
